// SingleMambaBlock_24223615549512
// MI455X (gfx1250) — matrix operand order repaired and checked by reading, not compiled, not run on hardware
//
#include <hip/hip_runtime.h>

// ---------- CDNA5 WMMA / TDM types ----------
typedef __attribute__((ext_vector_type(8)))  float        v8f;
typedef __attribute__((ext_vector_type(16))) _Float16     v16h;
typedef __attribute__((ext_vector_type(4)))  unsigned int v4u;
typedef __attribute__((ext_vector_type(8)))  int          v8i;
typedef __attribute__((ext_vector_type(4)))  int          v4i;

#ifndef __has_builtin
#define __has_builtin(x) 0
#endif
// tensor_load_to_lds exists on both toolchains but with different arity:
// ROCm 7.2 (clang-22): 5 args; amdgpu-toolchain (clang-23+): 6 args (extra int32x8).
#if __has_builtin(__builtin_amdgcn_tensor_load_to_lds) && \
    __has_builtin(__builtin_amdgcn_s_wait_tensorcnt)
#define USE_TDM 1
#else
#define USE_TDM 0
#endif

#define BB    2
#define DIMC  64
#define HH    128
#define WW    128
#define LL    (HH*WW)        // 16384
#define NST   8              // state size DS
#define CH    128            // scan chunk length
#define NCH   (LL/CH)        // 128 chunks

__device__ __forceinline__ float siluf(float v) { return v / (1.0f + __expf(-v)); }
__device__ __forceinline__ float softplusf(float v) { return v > 20.0f ? v : log1pf(__expf(v)); }

#if USE_TDM
// Issue one TDM descriptor: copy a 2D tile (tile_d0 x tile_d1 DWORDs, row stride
// `stride` DWORDs) from global memory to LDS.  D# layout per ISA 08 §8.3/8.4:
// group0 = {count=1, lds_addr, global_addr[56:0], type=2}; group1 = dims/strides,
// data_size=4B; groups 2/3 zero (<=2D tile).  Call from ONE wave only (uniform
// branch) -- TDM ignores EXEC and issues once per wave.
__device__ __forceinline__ void tdm_load_2d(const void* gsrc, void* lds_dst,
                                            unsigned tile_d0, unsigned tile_d1,
                                            unsigned long long stride_d0,
                                            unsigned td0, unsigned td1) {
  unsigned long long ga = (unsigned long long)(size_t)gsrc;
  unsigned lds = (unsigned)(size_t)lds_dst;             // generic LDS ptr: low 32 bits = LDS byte addr
  v4u g0 = { 1u,                                        // count=1, gather off
             lds,                                       // lds_addr
             (unsigned)ga,                              // global_addr[31:0]
             (unsigned)((ga >> 32) & 0x1FFFFFFu) | 0x80000000u }; // addr[56:32] | type=2
  v8i g1 = { (int)(2u << 16),                           // data_size = 4 bytes
             (int)((td0 & 0xffffu) << 16),              // tensor_dim0[15:0]
             (int)((td0 >> 16) | ((td1 & 0xffffu) << 16)),
             (int)(((td1 >> 16) & 0xffffu) | (tile_d0 << 16)),   // tile_dim0
             (int)(tile_d1 & 0xffffu),                  // tile_dim1 (tile_dim2=0)
             (int)(unsigned)stride_d0,                  // tensor_dim0_stride lo32
             (int)((unsigned)(stride_d0 >> 32) & 0xffffu),
             0 };
  v4i z = {0, 0, 0, 0};
#if __clang_major__ >= 23
  v8i z8 = {0, 0, 0, 0, 0, 0, 0, 0};
  __builtin_amdgcn_tensor_load_to_lds(g0, g1, z, z, z8, 0);   // 6-arg (clang-23+)
#else
  __builtin_amdgcn_tensor_load_to_lds(g0, g1, z, z, 0);       // 5-arg (ROCm 7.2)
#endif
}
#endif

// A fragment: 16x32 f16, per ISA 7.12.2 table (lanes 0-15: K pairs 0..7 then 16..23;
// lanes 16-31: K pairs 8..15 then 24..31). sA is a 16x64 row-major LDS tile, ko in {0,32}.
__device__ __forceinline__ v16h load_a_frag(const _Float16* sA, int lane, int ko) {
  int m  = lane & 15;
  int kb = (lane < 16) ? 0 : 8;
  v16h a;
#pragma unroll
  for (int j = 0; j < 8; ++j) {
    int k = ko + ((j >> 2) << 4) + kb + ((j & 3) << 1);
    a[2*j]   = sA[m*64 + k];
    a[2*j+1] = sA[m*64 + k + 1];
  }
  return a;
}

// B fragment: 32x16 f16 (K x N). Lane gives N=lane&15; same K order as A fragment
// (lanes 0-15: K pairs 0..7 then 16..23; lanes 16-31: K pairs 8..15 then 24..31).
// sW is weight W[n][k] row-major, row stride 64 (out = act @ W^T).
__device__ __forceinline__ v16h load_b_frag(const _Float16* sW, int lane, int n0, int ko) {
  int n  = lane & 15;
  int kh = (lane < 16) ? 0 : 8;
  v16h bf;
#pragma unroll
  for (int j = 0; j < 8; ++j) {
    int k = ko + ((j >> 2) << 4) + kh + ((j & 3) << 1);
    bf[2*j]   = sW[(n0+n)*64 + k];
    bf[2*j+1] = sW[(n0+n)*64 + k + 1];
  }
  return bf;
}

// ---------- K0: zero y_sum, convert GEMM weights to f16 ----------
__global__ void k_prep(const float* __restrict__ w_in, const float* __restrict__ w_out,
                       _Float16* __restrict__ w_in16, _Float16* __restrict__ w_out16,
                       float* __restrict__ y_sum) {
  int idx = blockIdx.x * blockDim.x + threadIdx.x;
  if (idx < BB*LL*DIMC) y_sum[idx] = 0.0f;
  if (idx < 2*DIMC*DIMC) w_in16[idx]  = (_Float16)w_in[idx];
  if (idx < DIMC*DIMC)   w_out16[idx] = (_Float16)w_out[idx];
}

// ---------- K1: LayerNorm + in_proj (WMMA f16, 64 -> 128) ----------
// TDM build: activation tile DMA'd as a true 2D tensor tile (16 l-values x 64
// channels, row stride L) into iT[c][m]; weight tile DMA'd 1D.  LN math overlaps
// the weight DMA; s_wait_tensorcnt(1) releases the tile, (0) gates the WMMAs.
__global__ void k_ln_inproj(const float* __restrict__ input, const float* __restrict__ ln_g,
                            const float* __restrict__ ln_b, const _Float16* __restrict__ w16,
                            float* __restrict__ xbuf, float* __restrict__ zbuf) {
  __shared__ float    iT[64*16];      // [channel c][m] (TDM tile layout)
  __shared__ _Float16 aT[16*64];
  __shared__ _Float16 wT[128*64];
  __shared__ float gS[64], bS[64], muS[16], rsS[16];
  int tid = threadIdx.x;              // 64 threads = 2 waves
  int b  = blockIdx.y;
  int l0 = blockIdx.x * 16;
#if USE_TDM
  if (tid < 32) {                     // wave 0 issues both DMAs (uniform branch)
    tdm_load_2d(input + (size_t)b*DIMC*LL + l0, iT, 16, 64, (unsigned long long)LL, LL, DIMC);
    tdm_load_2d(w16, wT, 4096, 1, 4096ull, 4096, 1);   // 16 KB contiguous
  }
#else
  {
    const float* inp = input + (size_t)b*DIMC*LL + (size_t)tid*LL + l0;   // channel tid
#pragma unroll
    for (int m = 0; m < 16; ++m) iT[tid*16 + m] = inp[m];
    for (int idx = tid; idx < 128*64; idx += 64) wT[idx] = w16[idx];
  }
#endif
  gS[tid] = ln_g[tid]; bS[tid] = ln_b[tid];
#if USE_TDM
  __builtin_amdgcn_s_wait_tensorcnt(1);   // first DMA (activation tile) complete
#endif
  __syncthreads();
  if (tid < 16) {
    float s = 0.f, s2 = 0.f;
    for (int c = 0; c < 64; ++c) { float v = iT[c*16 + tid]; s += v; s2 += v*v; }
    float mu  = s * (1.0f/64.0f);
    float var = s2 * (1.0f/64.0f) - mu*mu;
    muS[tid] = mu; rsS[tid] = rsqrtf(var + 1e-5f);
  }
  __syncthreads();
#pragma unroll
  for (int m = 0; m < 16; ++m)
    aT[m*64+tid] = (_Float16)((iT[tid*16 + m] - muS[m]) * rsS[m] * gS[tid] + bS[tid]);
#if USE_TDM
  __builtin_amdgcn_s_wait_tensorcnt(0);   // weight tile complete
#endif
  __syncthreads();
  int wave = tid >> 5, lane = tid & 31;
  v16h a0 = load_a_frag(aT, lane, 0);
  v16h a1 = load_a_frag(aT, lane, 32);
  int n = lane & 15, mh = (lane < 16) ? 0 : 8;
  for (int ti = 0; ti < 4; ++ti) {
    int n0 = (wave*4 + ti) * 16;
    v8f acc = {};
    v16h b0 = load_b_frag(wT, lane, n0, 0);
    acc = __builtin_amdgcn_wmma_f32_16x16x32_f16(false, a0, false, b0, (short)0, acc, false, false);
    v16h b1 = load_b_frag(wT, lane, n0, 32);
    acc = __builtin_amdgcn_wmma_f32_16x16x32_f16(false, a1, false, b1, (short)0, acc, false, false);
    int nout = n0 + n;
    float* dst = (nout < 64) ? xbuf : zbuf;
    int ncol = nout & 63;
#pragma unroll
    for (int j = 0; j < 8; ++j)
      dst[((size_t)b*LL + l0 + mh + j)*64 + ncol] = acc[j];
  }
}

// ---------- K2: direction gather + causal depthwise conv + SiLU ----------
__global__ void k_conv(const float* __restrict__ xbuf, const float* __restrict__ conv_w,
                       const float* __restrict__ conv_b, float* __restrict__ xd, int dir) {
  int gid = blockIdx.x * blockDim.x + threadIdx.x;
  int d = gid & 63;
  int q = gid >> 6;
  int b = blockIdx.y;
  int rev = dir & 1, trans = dir >> 1;
  float acc = conv_b[dir*64 + d];
  const float* w = conv_w + dir*256 + d*4;    // (NDIR, DIM, 1, 4)
#pragma unroll
  for (int j = 0; j < 4; ++j) {
    int tq = q - 3 + j;                       // causal pad (K-1,0)
    if (tq >= 0) {
      int t2 = rev ? (LL - 1 - tq) : tq;
      int s  = trans ? (((t2 & 127) << 7) | (t2 >> 7)) : t2;  // perm: (t%H)*W + t/H
      acc += w[j] * xbuf[((size_t)b*LL + s)*64 + d];
    }
  }
  xd[((size_t)b*LL + q)*64 + d] = siluf(acc);
}

// ---------- K2b: x_proj (64->20) + dt_proj + softplus ----------
__global__ void k_xproj(const float* __restrict__ xd, const float* __restrict__ xproj_w,
                        const float* __restrict__ dtproj_w, const float* __restrict__ dtproj_b,
                        float* __restrict__ delta, float* __restrict__ Bst, float* __restrict__ Cst,
                        int dir) {
  __shared__ float wS[20*64];
  __shared__ float dtwS[64*4];
  __shared__ float dtbS[64];
  int tid = threadIdx.x;                      // 256 threads
  for (int i = tid; i < 20*64; i += 256) wS[i] = xproj_w[dir*20*64 + i];
  dtwS[tid] = dtproj_w[dir*256 + tid];
  if (tid < 64) dtbS[tid] = dtproj_b[dir*64 + tid];
  __syncthreads();
  int row = blockIdx.x * 256 + tid;           // combined b*L + t
  float acc[20];
#pragma unroll
  for (int o = 0; o < 20; ++o) acc[o] = 0.f;
  const float* xr = xd + (size_t)row*64;
  for (int k = 0; k < 64; ++k) {
    float xv = xr[k];
#pragma unroll
    for (int o = 0; o < 20; ++o) acc[o] += xv * wS[o*64 + k];
  }
  float* drw = delta + (size_t)row*64;
  for (int dd = 0; dd < 64; ++dd) {
    float s = dtbS[dd];
#pragma unroll
    for (int r = 0; r < 4; ++r) s += acc[r] * dtwS[dd*4 + r];
    drw[dd] = softplusf(s);
  }
#pragma unroll
  for (int nn = 0; nn < NST; ++nn) {
    Bst[(size_t)row*8 + nn] = acc[4 + nn];
    Cst[(size_t)row*8 + nn] = acc[12 + nn];
  }
}

// ---------- K3a: per-chunk local scan (h_in = 0) ----------
__global__ void k_scan1(const float* __restrict__ delta, const float* __restrict__ xd,
                        const float* __restrict__ Bst, const float* __restrict__ A_log,
                        float* __restrict__ ch_h, float* __restrict__ ch_a, int dir) {
  __shared__ float Bs[CH*8];
  int d = threadIdx.x;                        // one thread per channel d
  int c = blockIdx.x;
  int b = blockIdx.y;
  int t0 = c * CH;
  for (int i = d; i < CH*8; i += 64) Bs[i] = Bst[((size_t)b*LL + t0)*8 + i];
  float A[NST];
#pragma unroll
  for (int nn = 0; nn < NST; ++nn) A[nn] = -__expf(A_log[dir*512 + d*8 + nn]);
  __syncthreads();
  float h[NST], aP[NST];
#pragma unroll
  for (int nn = 0; nn < NST; ++nn) { h[nn] = 0.f; aP[nn] = 1.f; }
  const float* dR = delta + ((size_t)b*LL + t0)*64 + d;
  const float* uR = xd    + ((size_t)b*LL + t0)*64 + d;
  for (int t = 0; t < CH; ++t) {
    __builtin_prefetch(dR + (t+16)*64, 0, 1);   // global_prefetch_b8, 16 steps ahead
    __builtin_prefetch(uR + (t+16)*64, 0, 1);
    float dt = dR[t*64];
    float du = dt * uR[t*64];
#pragma unroll
    for (int nn = 0; nn < NST; ++nn) {
      float dA = __expf(dt * A[nn]);
      aP[nn] *= dA;
      h[nn] = dA * h[nn] + du * Bs[t*8 + nn];
    }
  }
  size_t base = (((size_t)b*NCH + c)*64 + d)*8;
#pragma unroll
  for (int nn = 0; nn < NST; ++nn) { ch_h[base+nn] = h[nn]; ch_a[base+nn] = aP[nn]; }
}

// ---------- K3b: cross-chunk prefix ----------
__global__ void k_scan2(const float* __restrict__ ch_h, const float* __restrict__ ch_a,
                        float* __restrict__ pre) {
  int tid = threadIdx.x;                      // 512 = 64 channels x 8 states
  int b = blockIdx.x;
  float hin = 0.f;
  for (int c = 0; c < NCH; ++c) {
    size_t idx = ((size_t)b*NCH + c)*512 + tid;
    pre[idx] = hin;
    hin = ch_a[idx]*hin + ch_h[idx];
  }
}

// ---------- K3c: re-scan with prefix, emit y, scatter to original order ----------
__global__ void k_scan3(const float* __restrict__ delta, const float* __restrict__ xd,
                        const float* __restrict__ Bst, const float* __restrict__ Cst,
                        const float* __restrict__ A_log, const float* __restrict__ D_skip,
                        const float* __restrict__ pre, float* __restrict__ y_sum, int dir) {
  __shared__ float Bs[CH*8];
  __shared__ float Cs[CH*8];
  int d = threadIdx.x;
  int c = blockIdx.x;
  int b = blockIdx.y;
  int t0 = c * CH;
  for (int i = d; i < CH*8; i += 64) {
    Bs[i] = Bst[((size_t)b*LL + t0)*8 + i];
    Cs[i] = Cst[((size_t)b*LL + t0)*8 + i];
  }
  float A[NST], h[NST];
  size_t pbase = (((size_t)b*NCH + c)*64 + d)*8;
#pragma unroll
  for (int nn = 0; nn < NST; ++nn) {
    A[nn] = -__expf(A_log[dir*512 + d*8 + nn]);
    h[nn] = pre[pbase + nn];
  }
  float Dv = D_skip[dir*64 + d];
  __syncthreads();
  int rev = dir & 1, trans = dir >> 1;
  const float* dR = delta + ((size_t)b*LL + t0)*64 + d;
  const float* uR = xd    + ((size_t)b*LL + t0)*64 + d;
  for (int t = 0; t < CH; ++t) {
    __builtin_prefetch(dR + (t+16)*64, 0, 1);
    __builtin_prefetch(uR + (t+16)*64, 0, 1);
    float dt = dR[t*64];
    float u  = uR[t*64];
    float du = dt * u;
    float y = 0.f;
#pragma unroll
    for (int nn = 0; nn < NST; ++nn) {
      float dA = __expf(dt * A[nn]);
      h[nn] = dA * h[nn] + du * Bs[t*8 + nn];
      y += Cs[t*8 + nn] * h[nn];
    }
    y += u * Dv;
    int q  = t0 + t;
    int t2 = rev ? (LL - 1 - q) : q;
    int l  = trans ? (((t2 & 127) << 7) | (t2 >> 7)) : t2;   // inverse map = perm(t2)
    y_sum[((size_t)b*LL + l)*64 + d] += y;                   // one writer per (l,d) per dir
  }
}

// ---------- K4: LN + SiLU gate + out_proj (WMMA f16, 64 -> 64) + residual ----------
__global__ void k_final(const float* __restrict__ y_sum, const float* __restrict__ zbuf,
                        const float* __restrict__ on_g, const float* __restrict__ on_b,
                        const _Float16* __restrict__ w16, const float* __restrict__ input,
                        float* __restrict__ out) {
  __shared__ float    yT[16*64];
  __shared__ float    zT[16*64];
  __shared__ _Float16 aT[16*64];
  __shared__ _Float16 wT[64*64];
  __shared__ float gS[64], bS[64], muS[16], rsS[16];
  int tid = threadIdx.x;              // 64 threads = 2 waves
  int b  = blockIdx.y;
  int l0 = blockIdx.x * 16;
#if USE_TDM
  if (tid < 32) {                     // (l,d) tiles are contiguous 4 KB -> 1D DMAs
    tdm_load_2d(y_sum + ((size_t)b*LL + l0)*64, yT, 1024, 1, 1024ull, 1024, 1);
    tdm_load_2d(zbuf  + ((size_t)b*LL + l0)*64, zT, 1024, 1, 1024ull, 1024, 1);
    tdm_load_2d(w16, wT, 2048, 1, 2048ull, 2048, 1);   // 8 KB weights
  }
#else
#pragma unroll
  for (int m = 0; m < 16; ++m) {
    yT[m*64+tid] = y_sum[((size_t)b*LL + l0 + m)*64 + tid];
    zT[m*64+tid] = zbuf [((size_t)b*LL + l0 + m)*64 + tid];
  }
  for (int idx = tid; idx < 64*64; idx += 64) wT[idx] = w16[idx];
#endif
  gS[tid] = on_g[tid]; bS[tid] = on_b[tid];
#if USE_TDM
  __builtin_amdgcn_s_wait_tensorcnt(1);   // in-order: yT and zT DMAs complete
#endif
  __syncthreads();
  if (tid < 16) {
    float s = 0.f, s2 = 0.f;
    for (int c = 0; c < 64; ++c) { float v = yT[tid*64+c] * 0.25f; s += v; s2 += v*v; }
    float mu  = s * (1.0f/64.0f);
    float var = s2 * (1.0f/64.0f) - mu*mu;
    muS[tid] = mu; rsS[tid] = rsqrtf(var + 1e-5f);
  }
  __syncthreads();
#pragma unroll
  for (int m = 0; m < 16; ++m) {
    float v  = (yT[m*64+tid]*0.25f - muS[m]) * rsS[m] * gS[tid] + bS[tid];
    float zv = zT[m*64+tid];
    aT[m*64+tid] = (_Float16)(v * siluf(zv));
  }
#if USE_TDM
  __builtin_amdgcn_s_wait_tensorcnt(0);   // weight tile complete
#endif
  __syncthreads();
  int wave = tid >> 5, lane = tid & 31;
  v16h a0 = load_a_frag(aT, lane, 0);
  v16h a1 = load_a_frag(aT, lane, 32);
  int n = lane & 15, mh = (lane < 16) ? 0 : 8;
  for (int ti = 0; ti < 2; ++ti) {
    int n0 = (wave*2 + ti) * 16;
    v8f acc = {};
    v16h b0 = load_b_frag(wT, lane, n0, 0);
    acc = __builtin_amdgcn_wmma_f32_16x16x32_f16(false, a0, false, b0, (short)0, acc, false, false);
    v16h b1 = load_b_frag(wT, lane, n0, 32);
    acc = __builtin_amdgcn_wmma_f32_16x16x32_f16(false, a1, false, b1, (short)0, acc, false, false);
    int nout = n0 + n;
#pragma unroll
    for (int j = 0; j < 8; ++j) {
      size_t oi = (size_t)b*DIMC*LL + (size_t)nout*LL + (l0 + mh + j);
      out[oi] = acc[j] + input[oi];       // residual, transposed back to (B,C,H,W)
    }
  }
}

extern "C" void kernel_launch(void* const* d_in, const int* in_sizes, int n_in,
                              void* d_out, int out_size, void* d_ws, size_t ws_size,
                              hipStream_t stream) {
  (void)in_sizes; (void)n_in; (void)out_size; (void)ws_size;
  const float* input     = (const float*)d_in[0];
  const float* ln_g      = (const float*)d_in[1];
  const float* ln_b      = (const float*)d_in[2];
  const float* in_proj_w = (const float*)d_in[3];
  const float* conv_w    = (const float*)d_in[4];
  const float* conv_b    = (const float*)d_in[5];
  const float* xproj_w   = (const float*)d_in[6];
  const float* dtproj_w  = (const float*)d_in[7];
  const float* dtproj_b  = (const float*)d_in[8];
  const float* A_log     = (const float*)d_in[9];
  const float* D_skip    = (const float*)d_in[10];
  const float* on_g      = (const float*)d_in[11];
  const float* on_b      = (const float*)d_in[12];
  const float* out_proj_w= (const float*)d_in[13];
  float* out = (float*)d_out;

  char* ws = (char*)d_ws;
  size_t off = 0;
  auto alloc = [&](size_t bytes) {
    void* p = ws + off;
    off = (off + bytes + 255) & ~(size_t)255;
    return p;
  };
  const size_t ACT = sizeof(float)*(size_t)BB*LL*DIMC;     // 8 MiB
  float* xbuf   = (float*)alloc(ACT);
  float* zbuf   = (float*)alloc(ACT);
  float* y_sum  = (float*)alloc(ACT);
  float* xd     = (float*)alloc(ACT);                      // per-direction, reused
  float* delta  = (float*)alloc(ACT);                      // per-direction, reused
  float* Bst    = (float*)alloc(sizeof(float)*(size_t)BB*LL*8);
  float* Cst    = (float*)alloc(sizeof(float)*(size_t)BB*LL*8);
  float* ch_h   = (float*)alloc(sizeof(float)*(size_t)BB*NCH*DIMC*8);
  float* ch_a   = (float*)alloc(sizeof(float)*(size_t)BB*NCH*DIMC*8);
  float* pre    = (float*)alloc(sizeof(float)*(size_t)BB*NCH*DIMC*8);
  _Float16* w_in16  = (_Float16*)alloc(sizeof(_Float16)*2*DIMC*DIMC);
  _Float16* w_out16 = (_Float16*)alloc(sizeof(_Float16)*DIMC*DIMC);

  // K0: zero accumulator + f16 weight conversion
  k_prep<<<dim3((BB*LL*DIMC)/256), dim3(256), 0, stream>>>(in_proj_w, out_proj_w,
                                                           w_in16, w_out16, y_sum);
  // K1: LN + in_proj via WMMA (weights/activations staged by TDM when available)
  k_ln_inproj<<<dim3(LL/16, BB), dim3(64), 0, stream>>>(input, ln_g, ln_b, w_in16, xbuf, zbuf);

  for (int dir = 0; dir < 4; ++dir) {
    k_conv <<<dim3((LL*DIMC)/256, BB), dim3(256), 0, stream>>>(xbuf, conv_w, conv_b, xd, dir);
    k_xproj<<<dim3((BB*LL)/256),       dim3(256), 0, stream>>>(xd, xproj_w, dtproj_w, dtproj_b,
                                                               delta, Bst, Cst, dir);
    k_scan1<<<dim3(NCH, BB), dim3(64),  0, stream>>>(delta, xd, Bst, A_log, ch_h, ch_a, dir);
    k_scan2<<<dim3(BB),      dim3(512), 0, stream>>>(ch_h, ch_a, pre);
    k_scan3<<<dim3(NCH, BB), dim3(64),  0, stream>>>(delta, xd, Bst, Cst, A_log, D_skip,
                                                     pre, y_sum, dir);
  }
  // K4: LN + gate + out_proj via WMMA + residual
  k_final<<<dim3(LL/16, BB), dim3(64), 0, stream>>>(y_sum, zbuf, on_g, on_b, w_out16, input, out);
}
